// GCN_29781303231101
// MI455X (gfx1250) — matrix operand order repaired and checked by reading, not compiled, not run on hardware
//
#include <hip/hip_runtime.h>
#include <hip/hip_fp16.h>

typedef __attribute__((ext_vector_type(16))) _Float16 v16h;
typedef __attribute__((ext_vector_type(8)))  _Float16 v8h;
typedef __attribute__((ext_vector_type(8)))  float    v8f;
typedef __attribute__((ext_vector_type(4)))  float    v4f;

#define HD   256      // hidden width (== F_IN)
#define CCLS 10
#define BNEPS 1e-5f

// ---------------------------------------------------------------- utilities
__global__ void k_fill(float* p, float v, int n) {
    int i = blockIdx.x * blockDim.x + threadIdx.x;
    if (i < n) p[i] = v;
}

__global__ void k_deg(const int* __restrict__ dst, float* deg, int E) {
    int e = blockIdx.x * blockDim.x + threadIdx.x;
    if (e < E) atomicAdd(&deg[dst[e]], 1.0f);
}

__global__ void k_rsqrt(float* d, int n) {
    int i = blockIdx.x * blockDim.x + threadIdx.x;
    if (i < n) d[i] = rsqrtf(fmaxf(d[i], 1.0f));   // deg includes self-loop -> >= 1
}

// W[k][n] fp32  ->  Wt[n][k] f16  (so B fragments load contiguous along K)
__global__ void k_wt(const float* __restrict__ W, _Float16* __restrict__ Wt) {
    int idx = blockIdx.x * blockDim.x + threadIdx.x;
    if (idx >= HD * HD) return;
    int k = idx / HD, n = idx % HD;
    Wt[(size_t)n * HD + k] = (_Float16)W[(size_t)k * HD + n];
}

// ---------------------------------------------------------------- WMMA GEMM
// Y[M,256] = X[M,256] @ W (W supplied transposed+f16 as Wt[256][256]).
// Block = 256 threads = 8 waves. Wave w: rows [ (bx*8+w)*16, +16 ), cols [by*64, +64).
// A (16x32 f16) per ISA layout: lane m=lane&15; lanes<16 hold K {0..7,16..23},
// lanes>=16 hold K {8..15,24..31}. B (32x16 f16): lane n=lane&15; lanes<16 K {0..7,16..23},
// lanes>=16 K {8..15,24..31}. D: lane col=lane&15, vgpr r -> row r + 8*(lane>=16).
__global__ __launch_bounds__(256)
void k_gemm(const float* __restrict__ X, const _Float16* __restrict__ Wt,
            float* __restrict__ Y, int nRows) {
    const int wave     = threadIdx.x >> 5;
    const int lane     = threadIdx.x & 31;
    const int half_sel = lane >> 4;     // 0 | 1
    const int lid      = lane & 15;
    const int rowTile  = blockIdx.x * 8 + wave;
    if (rowTile * 16 >= nRows) return;  // nRows is a multiple of 16
    const int row      = rowTile * 16 + lid;
    const int colBase  = blockIdx.y * 64;

    v8f acc[4] = {};
    const float* xr    = X + (size_t)row * HD;
    const int    abase = half_sel * 8;
    const int    bbase = half_sel * 8;

    for (int kk = 0; kk < HD; kk += 32) {
        // ---- A fragment: two 32B fp32 runs, convert to f16 in-register
        const float* pa = xr + kk + abase;
        v4f u0 = *(const v4f*)(pa);
        v4f u1 = *(const v4f*)(pa + 4);
        v4f u2 = *(const v4f*)(pa + 16);
        v4f u3 = *(const v4f*)(pa + 20);
        v16h a;
#pragma unroll
        for (int e = 0; e < 4; ++e) {
            a[e]      = (_Float16)u0[e];
            a[4 + e]  = (_Float16)u1[e];
            a[8 + e]  = (_Float16)u2[e];
            a[12 + e] = (_Float16)u3[e];
        }
        // ---- 4 column tiles: B fragment = two 16B f16 runs; then WMMA
#pragma unroll
        for (int t = 0; t < 4; ++t) {
            const _Float16* pb = Wt + (size_t)(colBase + t * 16 + lid) * HD + kk + bbase;
            v8h b0 = *(const v8h*)(pb);
            v8h b1 = *(const v8h*)(pb + 16);
            v16h b;
#pragma unroll
            for (int e = 0; e < 8; ++e) { b[e] = b0[e]; b[8 + e] = b1[e]; }
            acc[t] = __builtin_amdgcn_wmma_f32_16x16x32_f16(
                false, a, false, b, (short)0, acc[t], false, false);
        }
    }
#pragma unroll
    for (int t = 0; t < 4; ++t) {
        const int n = colBase + t * 16 + lid;
#pragma unroll
        for (int r = 0; r < 8; ++r) {
            const int m = rowTile * 16 + r + 8 * half_sel;
            Y[(size_t)m * HD + n] = acc[t][r];
        }
    }
}

// ---------------------------------------------------------------- GCN aggregate
// out[i][:] = hw[i][:] * dis[i]^2 + bias  (self-loop term; full overwrite init)
__global__ void k_selfloop(const float* __restrict__ hw, const float* __restrict__ dis,
                           const float* __restrict__ bias, float* __restrict__ out, int n) {
    long long idx = (long long)blockIdx.x * blockDim.x + threadIdx.x;  // n * 64 lanes
    if (idx >= (long long)n * (HD / 4)) return;
    int i = (int)(idx >> 6), c4 = (int)idx & 63;
    float s = dis[i]; s = s * s;
    v4f h = *(const v4f*)(hw + (size_t)i * HD + c4 * 4);
    v4f b = *(const v4f*)(bias + c4 * 4);
    v4f o;
#pragma unroll
    for (int e = 0; e < 4; ++e) o[e] = h[e] * s + b[e];
    *(v4f*)(out + (size_t)i * HD + c4 * 4) = o;
}

// out[dst] += hw[src] * dis[src]*dis[dst]; 64 threads per edge, float4 gathers
__global__ void k_edges(const float* __restrict__ hw, const float* __restrict__ dis,
                        const int* __restrict__ src, const int* __restrict__ dst,
                        float* __restrict__ out, int E) {
    long long tid = (long long)blockIdx.x * blockDim.x + threadIdx.x;
    int e = (int)(tid >> 6);
    if (e >= E) return;
    int lane = (int)tid & 63;
    int s = src[e], d = dst[e];
    float nrm = dis[s] * dis[d];
    v4f v = *(const v4f*)(hw + (size_t)s * HD + lane * 4);
    float* op = out + (size_t)d * HD + lane * 4;
    atomicAdd(op + 0, v[0] * nrm);
    atomicAdd(op + 1, v[1] * nrm);
    atomicAdd(op + 2, v[2] * nrm);
    atomicAdd(op + 3, v[3] * nrm);
}

// ---------------------------------------------------------------- BatchNorm
__global__ void k_bnstats(const float* __restrict__ y, float* sums, float* sqs, int n) {
    int c = threadIdx.x;  // 256 channels, coalesced column sweep
    float s = 0.f, q = 0.f;
    for (int i = blockIdx.x; i < n; i += gridDim.x) {
        float v = y[(size_t)i * HD + c];
        s += v; q += v * v;
    }
    atomicAdd(&sums[c], s);
    atomicAdd(&sqs[c], q);
}

__global__ void k_bnfin(const float* sums, const float* sqs,
                        const float* __restrict__ g, const float* __restrict__ be,
                        float* scale, float* shift, int n) {
    int c = threadIdx.x;
    float mean = sums[c] / (float)n;
    float var  = sqs[c] / (float)n - mean * mean;
    float inv  = rsqrtf(var + BNEPS);
    float sc   = inv * g[c];
    scale[c] = sc;
    shift[c] = be[c] - mean * sc;
}

__global__ void k_bnrelu(float* y, const float* __restrict__ scale,
                         const float* __restrict__ shift, int n) {
    long long idx = (long long)blockIdx.x * blockDim.x + threadIdx.x;
    if (idx >= (long long)n * HD) return;
    int c = (int)idx & (HD - 1);
    float v = y[idx] * scale[c] + shift[c];
    y[idx] = fmaxf(v, 0.f);
}

// ---------------------------------------------------------------- pooling + head
__global__ void k_cnt(const int* __restrict__ batch, float* cnt, int n) {
    int i = blockIdx.x * blockDim.x + threadIdx.x;
    if (i < n) atomicAdd(&cnt[batch[i]], 1.0f);
}

__global__ void k_pool(const float* __restrict__ y, const int* __restrict__ batch,
                       float* pooled, int n) {
    long long idx = (long long)blockIdx.x * blockDim.x + threadIdx.x;
    if (idx >= (long long)n * 64) return;
    int i = (int)(idx >> 6), c4 = (int)idx & 63;
    int g = batch[i];
    v4f v = *(const v4f*)(y + (size_t)i * HD + c4 * 4);
    float* p = pooled + (size_t)g * HD + c4 * 4;
    atomicAdd(p + 0, v[0]); atomicAdd(p + 1, v[1]);
    atomicAdd(p + 2, v[2]); atomicAdd(p + 3, v[3]);
}

__global__ void k_pooldiv(float* pooled, const float* cnt, int G) {
    int idx = blockIdx.x * blockDim.x + threadIdx.x;
    if (idx >= G * HD) return;
    pooled[idx] /= fmaxf(cnt[idx >> 8], 1.0f);
}

__global__ void k_lin1(const float* __restrict__ pooled, const float* __restrict__ w,
                       const float* __restrict__ b, float* z, int G) {
    int idx = blockIdx.x * blockDim.x + threadIdx.x;
    if (idx >= G * HD) return;
    int g = idx >> 8, c = idx & (HD - 1);
    const float* pr = pooled + (size_t)g * HD;
    float s = b[c];
    for (int k = 0; k < HD; ++k) s += pr[k] * w[(size_t)k * HD + c];
    z[idx] = s;
}

// BN over G rows + ReLU, single block (each thread owns one channel)
__global__ void k_bn1(float* z, const float* __restrict__ g1,
                      const float* __restrict__ b1, int G) {
    int c = threadIdx.x;
    float s = 0.f, q = 0.f;
    for (int g = 0; g < G; ++g) { float v = z[(size_t)g * HD + c]; s += v; q += v * v; }
    float mean = s / (float)G;
    float var  = q / (float)G - mean * mean;
    float inv  = rsqrtf(var + BNEPS);
    float sc = inv * g1[c], sh = b1[c] - mean * sc;
    for (int g = 0; g < G; ++g) {
        float v = z[(size_t)g * HD + c] * sc + sh;
        z[(size_t)g * HD + c] = fmaxf(v, 0.f);
    }
}

__global__ void k_lin2(const float* __restrict__ z, const float* __restrict__ w,
                       const float* __restrict__ b, float* out, int G) {
    int idx = blockIdx.x * blockDim.x + threadIdx.x;
    if (idx >= G * CCLS) return;
    int g = idx / CCLS, c = idx % CCLS;
    float s = b[c];
    for (int k = 0; k < HD; ++k) s += z[(size_t)g * HD + k] * w[(size_t)k * CCLS + c];
    out[idx] = s;
}

// ---------------------------------------------------------------- launcher
extern "C" void kernel_launch(void* const* d_in, const int* in_sizes, int n_in,
                              void* d_out, int out_size, void* d_ws, size_t ws_size,
                              hipStream_t stream) {
    const float* x     = (const float*)d_in[0];
    const int*   ei    = (const int*)d_in[1];
    const int*   batch = (const int*)d_in[2];
    const float* W[3]  = {(const float*)d_in[3], (const float*)d_in[7],  (const float*)d_in[11]};
    const float* bb[3] = {(const float*)d_in[4], (const float*)d_in[8],  (const float*)d_in[12]};
    const float* gg[3] = {(const float*)d_in[5], (const float*)d_in[9],  (const float*)d_in[13]};
    const float* be[3] = {(const float*)d_in[6], (const float*)d_in[10], (const float*)d_in[14]};
    const float* lin1_w = (const float*)d_in[15];
    const float* lin1_b = (const float*)d_in[16];
    const float* bn1_g  = (const float*)d_in[17];
    const float* bn1_b  = (const float*)d_in[18];
    const float* lin2_w = (const float*)d_in[19];
    const float* lin2_b = (const float*)d_in[20];

    const int N = in_sizes[0] / HD;
    const int E = in_sizes[1] / 2;
    const int G = out_size / CCLS;
    const int* src = ei;
    const int* dst = ei + E;

    // bump allocator on d_ws
    char* wp = (char*)d_ws;
    auto alloc = [&](size_t bytes) -> void* {
        void* p = wp;
        wp += (bytes + 255) & ~(size_t)255;
        return p;
    };
    float*    hw     = (float*)alloc((size_t)N * HD * 4);   // GEMM output
    float*    hbuf   = (float*)alloc((size_t)N * HD * 4);   // aggregate / activations
    float*    dis    = (float*)alloc((size_t)N * 4);        // deg^{-1/2}
    _Float16* Wt     = (_Float16*)alloc((size_t)HD * HD * 2);
    float*    sums   = (float*)alloc(512 * 4);              // [0:256)=sum, [256:512)=sumsq
    float*    sqs    = sums + 256;
    float*    scale  = (float*)alloc(HD * 4);
    float*    shift  = (float*)alloc(HD * 4);
    float*    pooled = (float*)alloc((size_t)G * HD * 4);
    float*    cnt    = (float*)alloc((size_t)G * 4);
    float*    z1     = (float*)alloc((size_t)G * HD * 4);

    // degrees -> dis (self-loop included)
    k_fill <<<(N + 255) / 256, 256, 0, stream>>>(dis, 1.0f, N);
    k_deg  <<<(E + 255) / 256, 256, 0, stream>>>(dst, dis, E);
    k_rsqrt<<<(N + 255) / 256, 256, 0, stream>>>(dis, N);

    const long long edgeThreads = (long long)E * 64;
    const long long elemThreads = (long long)N * HD;
    const float* layer_in = x;
    for (int l = 0; l < 3; ++l) {
        k_wt<<<(HD * HD + 255) / 256, 256, 0, stream>>>(W[l], Wt);
        dim3 ggrid((N / 16 + 7) / 8, HD / 64);
        k_gemm<<<ggrid, 256, 0, stream>>>(layer_in, Wt, hw, N);
        k_selfloop<<<(unsigned)((elemThreads / 4 + 255) / 256), 256, 0, stream>>>(hw, dis, bb[l], hbuf, N);
        k_edges<<<(unsigned)((edgeThreads + 255) / 256), 256, 0, stream>>>(hw, dis, src, dst, hbuf, E);
        k_fill<<<2, 256, 0, stream>>>(sums, 0.0f, 512);
        k_bnstats<<<1024, 256, 0, stream>>>(hbuf, sums, sqs, N);
        k_bnfin<<<1, 256, 0, stream>>>(sums, sqs, gg[l], be[l], scale, shift, N);
        k_bnrelu<<<(unsigned)((elemThreads + 255) / 256), 256, 0, stream>>>(hbuf, scale, shift, N);
        layer_in = hbuf;
    }

    // global mean pool
    k_fill<<<(G * HD + 255) / 256, 256, 0, stream>>>(pooled, 0.0f, G * HD);
    k_fill<<<1, 256, 0, stream>>>(cnt, 0.0f, G);
    k_cnt <<<(N + 255) / 256, 256, 0, stream>>>(batch, cnt, N);
    k_pool<<<(unsigned)(((long long)N * 64 + 255) / 256), 256, 0, stream>>>(hbuf, batch, pooled, N);
    k_pooldiv<<<(G * HD + 255) / 256, 256, 0, stream>>>(pooled, cnt, G);

    // MLP head (tiny; fp32 VALU)
    k_lin1<<<(G * HD + 255) / 256, 256, 0, stream>>>(pooled, lin1_w, lin1_b, z1, G);
    k_bn1 <<<1, 256, 0, stream>>>(z1, bn1_g, bn1_b, G);
    k_lin2<<<(G * CCLS + 255) / 256, 256, 0, stream>>>(z1, lin2_w, lin2_b, (float*)d_out, G);
}
